// LabelCorrelationHead_20736102105299
// MI455X (gfx1250) — matrix operand order repaired and checked by reading, not compiled, not run on hardware
//
#include <hip/hip_runtime.h>
#include <hip/hip_bf16.h>

#define BATCH   16384
#define FEAT    2048
#define NCLS    14
#define ADIM    64
#define QKDIM   (NCLS * 2 * ADIM)   // 1792
#define BN_EPS  1e-5f

typedef __attribute__((ext_vector_type(16))) _Float16 v16h;
typedef __attribute__((ext_vector_type(8)))  _Float16 v8h;
typedef __attribute__((ext_vector_type(4)))  _Float16 v4h;
typedef __attribute__((ext_vector_type(8)))  float    v8f;
typedef __attribute__((ext_vector_type(4)))  float    v4f;
typedef __attribute__((ext_vector_type(4)))  unsigned int v4u;
typedef __attribute__((ext_vector_type(4)))  int      v4i;
typedef __attribute__((ext_vector_type(8)))  int      v8i_t;

#if __has_builtin(__builtin_amdgcn_tensor_load_to_lds) && \
    __has_builtin(__builtin_amdgcn_s_wait_tensorcnt)
#define TDM_OK 1
#else
#define TDM_OK 0
#endif

// Build a v16h WMMA operand from two 16-byte chunks `gap` halfs apart.
// A-frag (16x32, lane = {m=lane&15, hf=lane>>4}): chunks at K = hf*8 and 16+hf*8  -> gap 16
// B-frag (32x16, lane = {n=lane&15, hf=lane>>4}): chunks at K = hf*8 and 16+hf*8  -> gap 16
static __device__ inline v16h make_frag(const _Float16* p, int gap) {
  v8h lo = *(const v8h*)p;
  v8h hi = *(const v8h*)(p + gap);
  v16h r;
#pragma unroll
  for (int i = 0; i < 8; ++i) { r[i] = lo[i]; r[i + 8] = hi[i]; }
  return r;
}

#if TDM_OK
// ---------------------------------------------------------------------------
// Tensor Data Mover: DMA one [tile_rows x 32-half] tile (row stride K halfs in
// global) into LDS at byte offset lds_off, with LDS padding of 4 DWORDs after
// every 16 DWORDs -> padded row stride of 40 halfs (bank-conflict-free
// ds_read_b128 fragment loads). D# layout per CDNA5 ISA ch.8.
// ---------------------------------------------------------------------------
static __device__ inline void tdm_load_tile(const _Float16* gptr, unsigned lds_off,
                                            unsigned tile_rows, unsigned tensor_rows,
                                            unsigned kdim) {
  unsigned long long ga = (unsigned long long)(uintptr_t)gptr;
  v4u g0;
  g0[0] = 1u;                                            // count=1, user mode
  g0[1] = lds_off;                                       // lds_addr [63:32]
  g0[2] = (unsigned)(ga & 0xFFFFFFFFull);                // global_addr[31:0]
  g0[3] = (unsigned)((ga >> 32) & 0x01FFFFFFull)         // global_addr[56:32]
          | (2u << 30);                                  // type=2 ("image")
  v8i_t g1;
  g1[0] = (int)((1u << 16)        // data_size = 1 -> 2 bytes
              | (1u << 20)        // pad_enable
              | (3u << 22)        // pad_interval code 3 -> every 16 DWORDs (64B row)
              | (3u << 25));      // pad_amount  code 3 -> 4 DWORDs (16B)
  g1[1] = (int)((kdim & 0xFFFFu) << 16);                        // tensor_dim0 lo
  g1[2] = (int)((kdim >> 16) | ((tensor_rows & 0xFFFFu) << 16));// dim0 hi | dim1 lo
  g1[3] = (int)((tensor_rows >> 16) | (32u << 16));             // dim1 hi | tile_dim0=32
  g1[4] = (int)(tile_rows & 0xFFFFu);                           // tile_dim1 (tile_dim2=0)
  g1[5] = (int)kdim;                                            // tensor_dim0_stride lo32
  g1[6] = 0;                                                    // stride hi | dim1_stride lo
  g1[7] = 0;
  v4i  z4 = {0, 0, 0, 0};
#if __clang_major__ >= 23
  v8i_t z8 = {0, 0, 0, 0, 0, 0, 0, 0};
  __builtin_amdgcn_tensor_load_to_lds(g0, g1, z4, z4, z8, 0);
#else
  __builtin_amdgcn_tensor_load_to_lds(g0, g1, z4, z4, 0);
#endif
}
#endif  // TDM_OK

// ---------------------------------------------------------------------------
// Utility kernels
// ---------------------------------------------------------------------------
__global__ void zero_f32_kernel(float* __restrict__ p, int n) {
  int i = blockIdx.x * blockDim.x + threadIdx.x;
  if (i < n) p[i] = 0.f;
}

__global__ void cvt_f16_kernel(const float* __restrict__ s, _Float16* __restrict__ d, size_t n) {
  size_t i = ((size_t)blockIdx.x * blockDim.x + threadIdx.x) * 4;
  if (i >= n) return;
  v4f v = *(const v4f*)(s + i);
  v4h o;
#pragma unroll
  for (int j = 0; j < 4; ++j) o[j] = (_Float16)v[j];
  *(v4h*)(d + i) = o;
}

// W_fc [14][FEAT] f32 -> [16][FEAT] f16, rows 14,15 zero-padded
__global__ void cvt_wfc_kernel(const float* __restrict__ s, _Float16* __restrict__ d) {
  int i = blockIdx.x * blockDim.x + threadIdx.x;   // < 16*FEAT
  int row = i >> 11;                               // FEAT == 2048
  d[i] = (row < NCLS) ? (_Float16)s[i] : (_Float16)0.f;
}

// ---------------------------------------------------------------------------
// Main WMMA GEMM: Out[M][N](f16) = A[M][K](f16) @ W[N][K]^T (f16)  [+bias, +stats]
// Block tile 128x64, 8 waves (4 M x 2 N), each wave 2x2 of 16x16 WMMA tiles.
// Tile staging: TDM double-buffered (wave 0 issues tensor_load_to_lds for
// K-step t+1, s_wait_tensorcnt(2) guarantees step t landed, barrier, compute).
// EPI==1: add f32 bias and accumulate per-column sum / sum-of-squares.
// ---------------------------------------------------------------------------
template <int EPI>
__global__ __launch_bounds__(256) void wmma_gemm_kernel(
    const _Float16* __restrict__ A,
    const _Float16* __restrict__ W,
    const float*    __restrict__ bias,
    _Float16*       __restrict__ Out,
    float* __restrict__ colsum, float* __restrict__ colsumsq,
    int M, int N, int K)
{
  constexpr int LDT = 40;                 // padded stride (halfs)
  __shared__ _Float16 ldsA[2][128 * LDT];
  __shared__ _Float16 ldsB[2][64 * LDT];
  __shared__ float ssum[64], ssq[64];

  const int tid  = threadIdx.x;
  const int lane = tid & 31;
  const int ln   = lane & 15;
  const int hf   = lane >> 4;
  const int wave = tid >> 5;
  const int wm   = wave >> 1;             // 0..3 (M strip of 32 rows)
  const int wn   = wave & 1;              // 0..1 (N strip of 32 cols)
  const int gm0  = blockIdx.y * 128;
  const int gn0  = blockIdx.x * 64;

  if (EPI) { if (tid < 64) { ssum[tid] = 0.f; ssq[tid] = 0.f; } }

  v8f acc[2][2] = {};

  const int KSTEPS = K / 32;

#if TDM_OK
  const unsigned ldsA_off = (unsigned)(uintptr_t)&ldsA[0][0];
  const unsigned ldsB_off = (unsigned)(uintptr_t)&ldsB[0][0];
  constexpr unsigned ABUF = 128 * LDT * 2;   // bytes per A buffer
  constexpr unsigned BBUF = 64 * LDT * 2;    // bytes per B buffer
  if (wave == 0) {   // prologue: stage K-step 0 into buffer 0
    tdm_load_tile(A + (size_t)gm0 * K, ldsA_off, 128u, (unsigned)M, (unsigned)K);
    tdm_load_tile(W + (size_t)gn0 * K, ldsB_off, 64u,  (unsigned)N, (unsigned)K);
  }
#else
  const int crow = tid >> 2;              // 0..63
  const int coff = (tid & 3) * 8;         // 0,8,16,24 halfs
#endif

  for (int t = 0; t < KSTEPS; ++t) {
#if TDM_OK
    const int cur = t & 1;
    if (wave == 0) {
      if (t + 1 < KSTEPS) {
        const int nxt = cur ^ 1;
        tdm_load_tile(A + (size_t)gm0 * K + (size_t)(t + 1) * 32, ldsA_off + nxt * ABUF,
                      128u, (unsigned)M, (unsigned)K);
        tdm_load_tile(W + (size_t)gn0 * K + (size_t)(t + 1) * 32, ldsB_off + nxt * BBUF,
                      64u, (unsigned)N, (unsigned)K);
        __builtin_amdgcn_s_wait_tensorcnt(2);   // pair t complete (in-order TDM)
      } else {
        __builtin_amdgcn_s_wait_tensorcnt(0);
      }
    }
    __syncthreads();                             // tile t visible to all waves
#else
    const int cur = 0;
    const int k0 = t * 32;
    __syncthreads();
#pragma unroll
    for (int p = 0; p < 2; ++p) {
      const int row = crow + p * 64;
      v8h v = *(const v8h*)(A + (size_t)(gm0 + row) * K + (k0 + coff));
      *(v8h*)(&ldsA[0][0] + row * LDT + coff) = v;
    }
    {
      v8h v = *(const v8h*)(W + (size_t)(gn0 + crow) * K + (k0 + coff));
      *(v8h*)(&ldsB[0][0] + crow * LDT + coff) = v;
    }
    if (k0 + 32 < K) {
      __builtin_prefetch(A + (size_t)(gm0 + crow) * K + (k0 + 32 + coff), 0, 0);
      __builtin_prefetch(W + (size_t)(gn0 + crow) * K + (k0 + 32 + coff), 0, 0);
    }
    __syncthreads();
#endif

    v16h af[2], bf[2];
#pragma unroll
    for (int mi = 0; mi < 2; ++mi)
      af[mi] = make_frag(&ldsA[cur][0] + (wm * 32 + mi * 16 + ln) * LDT + hf * 8, 16);
#pragma unroll
    for (int ni = 0; ni < 2; ++ni)
      bf[ni] = make_frag(&ldsB[cur][0] + (wn * 32 + ni * 16 + ln) * LDT + hf * 8, 16);

#pragma unroll
    for (int mi = 0; mi < 2; ++mi)
#pragma unroll
      for (int ni = 0; ni < 2; ++ni)
        acc[mi][ni] = __builtin_amdgcn_wmma_f32_16x16x32_f16(
            false, af[mi], false, bf[ni], (short)0, acc[mi][ni], false, false);

    __syncthreads();                             // protect buffer reuse next iter
  }

  // Epilogue: C/D layout -> lane holds col = ln, rows hf*8 .. hf*8+7
#pragma unroll
  for (int mi = 0; mi < 2; ++mi) {
#pragma unroll
    for (int ni = 0; ni < 2; ++ni) {
      const int lcol = wn * 32 + ni * 16 + ln;
      const int col  = gn0 + lcol;
      const int row0 = gm0 + wm * 32 + mi * 16 + hf * 8;
      const float bb = EPI ? bias[col] : 0.f;
      float s1 = 0.f, s2 = 0.f;
#pragma unroll
      for (int r = 0; r < 8; ++r) {
        float v = acc[mi][ni][r] + bb;
        if (EPI) { s1 += v; s2 += v * v; }
        Out[(size_t)(row0 + r) * N + col] = (_Float16)v;
      }
      if (EPI) { atomicAdd(&ssum[lcol], s1); atomicAdd(&ssq[lcol], s2); }
    }
  }
  if (EPI) {
    __syncthreads();
    if (tid < 64) {
      atomicAdd(&colsum[gn0 + tid],   ssum[tid]);
      atomicAdd(&colsumsq[gn0 + tid], ssq[tid]);
    }
  }
}

// ---------------------------------------------------------------------------
// BN stats -> affine scale/shift
// ---------------------------------------------------------------------------
__global__ void finalize_stats_kernel(const float* __restrict__ colsum,
                                      const float* __restrict__ colsumsq,
                                      const float* __restrict__ bn_gamma,
                                      const float* __restrict__ bn_beta,
                                      float* __restrict__ scale,
                                      float* __restrict__ shift, int F, float invB) {
  int i = blockIdx.x * blockDim.x + threadIdx.x;
  if (i >= F) return;
  float mu   = colsum[i] * invB;
  float var  = colsumsq[i] * invB - mu * mu;   // biased variance, like torch BN
  float rstd = rsqrtf(var + BN_EPS);
  float sc   = rstd * bn_gamma[i];
  scale[i] = sc;
  shift[i] = bn_beta[i] - mu * sc;
}

// h16 <- relu(h16 * scale + shift), in place, 8 halfs per thread
__global__ __launch_bounds__(256) void norm_relu_kernel(_Float16* __restrict__ h,
                                                        const float* __restrict__ scale,
                                                        const float* __restrict__ shift,
                                                        size_t total) {
  size_t idx = ((size_t)blockIdx.x * blockDim.x + threadIdx.x) * 8;
  if (idx >= total) return;
  int col = (int)(idx & (FEAT - 1));
  v8h v = *(const v8h*)(h + idx);
  v8h o;
#pragma unroll
  for (int i = 0; i < 8; ++i) {
    float x = (float)v[i] * scale[col + i] + shift[col + i];
    o[i] = (_Float16)fmaxf(x, 0.f);
  }
  *(v8h*)(h + idx) = o;
}

// ---------------------------------------------------------------------------
// Logits GEMM: Lg[M][16](f32) = A[M][K] @ Wfc[16][K]^T + b_fc (cols>=14 padded 0)
// One 16x16 WMMA tile per wave, operands fetched straight from global (Wfc is
// 64 KB f16 -> L2 resident).
// ---------------------------------------------------------------------------
__global__ __launch_bounds__(256) void wmma_logits_kernel(
    const _Float16* __restrict__ A,
    const _Float16* __restrict__ Wfc,
    const float*    __restrict__ bfc,
    float*          __restrict__ Lg,
    int M, int K)
{
  const int tid  = threadIdx.x;
  const int lane = tid & 31;
  const int ln   = lane & 15;
  const int hf   = lane >> 4;
  const int wave = tid >> 5;
  const int gm0  = blockIdx.x * 128 + wave * 16;

  v8f acc = {};
  for (int k0 = 0; k0 < K; k0 += 32) {
    const _Float16* pa = A   + (size_t)(gm0 + ln) * K + k0 + hf * 8;
    const _Float16* pb = Wfc + (size_t)ln        * K + k0 + hf * 8;
    v16h a = make_frag(pa, 16);
    v16h b = make_frag(pb, 16);
    acc = __builtin_amdgcn_wmma_f32_16x16x32_f16(false, a, false, b, (short)0, acc,
                                                 false, false);
  }
  const float bb = (ln < NCLS) ? bfc[ln] : 0.f;
#pragma unroll
  for (int r = 0; r < 8; ++r)
    Lg[(size_t)(gm0 + hf * 8 + r) * 16 + ln] = acc[r] + bb;
}

// ---------------------------------------------------------------------------
// Per-row label-correlation head: attn = q k^T, diag subtract, max-abs
// normalize, corr = attn @ sigmoid(logits); out = logits + gamma * corr.
// One block per batch row.
// ---------------------------------------------------------------------------
__global__ __launch_bounds__(256) void attn_head_kernel(
    const _Float16* __restrict__ qk,     // [B][QKDIM]  (q | k)
    const float*    __restrict__ Lg,     // [B][16]
    const float*    __restrict__ gamma,  // [NCLS]
    float*          __restrict__ out)    // [B][NCLS]
{
  __shared__ _Float16 sqk[QKDIM];
  __shared__ float sattn[NCLS * NCLS];
  __shared__ float ssig[NCLS];
  __shared__ float slog[NCLS];

  const int b   = blockIdx.x;
  const int tid = threadIdx.x;

  for (int i = tid; i < QKDIM; i += 256) sqk[i] = qk[(size_t)b * QKDIM + i];
  if (tid < NCLS) {
    float l = Lg[(size_t)b * 16 + tid];
    slog[tid] = l;
    ssig[tid] = 1.f / (1.f + expf(-l));
  }
  __syncthreads();

  if (tid < NCLS * NCLS) {
    const int c = tid / NCLS, e = tid % NCLS;
    float d = 0.f;
#pragma unroll
    for (int i = 0; i < ADIM; ++i)
      d += (float)sqk[c * ADIM + i] * (float)sqk[NCLS * ADIM + e * ADIM + i];
    sattn[tid] = d;
  }
  __syncthreads();

  if (tid < NCLS) {
    const int c = tid;
    const float diag = sattn[c * NCLS + c];
    float m = 0.f;
#pragma unroll
    for (int e = 0; e < NCLS; ++e) m = fmaxf(m, fabsf(sattn[c * NCLS + e] - diag));
    float corr = 0.f;
#pragma unroll
    for (int e = 0; e < NCLS; ++e) corr += (sattn[c * NCLS + e] - diag) / m * ssig[e];
    out[(size_t)b * NCLS + c] = slog[c] + gamma[c] * corr;
  }
}

// ---------------------------------------------------------------------------
// Host side
// ---------------------------------------------------------------------------
extern "C" void kernel_launch(void* const* d_in, const int* in_sizes, int n_in,
                              void* d_out, int out_size, void* d_ws, size_t ws_size,
                              hipStream_t stream) {
  const float* features = (const float*)d_in[0];
  const float* W_proj   = (const float*)d_in[1];
  const float* b_proj   = (const float*)d_in[2];
  const float* bn_gamma = (const float*)d_in[3];
  const float* bn_beta  = (const float*)d_in[4];
  const float* W_qk     = (const float*)d_in[5];
  const float* gamma    = (const float*)d_in[6];
  const float* W_fc     = (const float*)d_in[7];
  const float* b_fc     = (const float*)d_in[8];
  float* out = (float*)d_out;

  char* ws = (char*)d_ws;
  auto take = [&](size_t bytes) -> char* {
    char* p = ws;
    ws += (bytes + 255) & ~(size_t)255;
    return p;
  };
  _Float16* feat16 = (_Float16*)take((size_t)BATCH * FEAT * 2);
  _Float16* h16    = (_Float16*)take((size_t)BATCH * FEAT * 2);
  _Float16* qk16   = (_Float16*)take((size_t)BATCH * QKDIM * 2);
  _Float16* wp16   = (_Float16*)take((size_t)FEAT * FEAT * 2);
  _Float16* wqk16  = (_Float16*)take((size_t)QKDIM * FEAT * 2);
  _Float16* wfc16  = (_Float16*)take((size_t)16 * FEAT * 2);
  float* logits32  = (float*)take((size_t)BATCH * 16 * 4);
  float* colsum    = (float*)take((size_t)FEAT * 4);   // contiguous with colsumsq
  float* colsumsq  = (float*)take((size_t)FEAT * 4);
  float* bnscale   = (float*)take((size_t)FEAT * 4);
  float* bnshift   = (float*)take((size_t)FEAT * 4);
  (void)ws_size; (void)in_sizes; (void)n_in; (void)out_size;

  const dim3 blk(256);

  // 0) zero BN stat accumulators (colsum & colsumsq are adjacent)
  zero_f32_kernel<<<dim3((2 * FEAT + 255) / 256), blk, 0, stream>>>(colsum, 2 * FEAT);

  // 1) f32 -> f16 conversions
  {
    size_t n = (size_t)BATCH * FEAT;
    cvt_f16_kernel<<<dim3((unsigned)((n / 4 + 255) / 256)), blk, 0, stream>>>(features, feat16, n);
  }
  {
    size_t n = (size_t)FEAT * FEAT;
    cvt_f16_kernel<<<dim3((unsigned)((n / 4 + 255) / 256)), blk, 0, stream>>>(W_proj, wp16, n);
  }
  {
    size_t n = (size_t)QKDIM * FEAT;
    cvt_f16_kernel<<<dim3((unsigned)((n / 4 + 255) / 256)), blk, 0, stream>>>(W_qk, wqk16, n);
  }
  cvt_wfc_kernel<<<dim3(16 * FEAT / 256), blk, 0, stream>>>(W_fc, wfc16);

  // 2) GEMM1: h = features @ W_proj^T + b_proj, with fused BN stat accumulation
  wmma_gemm_kernel<1><<<dim3(FEAT / 64, BATCH / 128), blk, 0, stream>>>(
      feat16, wp16, b_proj, h16, colsum, colsumsq, BATCH, FEAT, FEAT);

  // 3) BN finalize + normalize/ReLU in place
  finalize_stats_kernel<<<dim3((FEAT + 255) / 256), blk, 0, stream>>>(
      colsum, colsumsq, bn_gamma, bn_beta, bnscale, bnshift, FEAT, 1.f / (float)BATCH);
  norm_relu_kernel<<<dim3((unsigned)((size_t)BATCH * FEAT / 8 / 256)), blk, 0, stream>>>(
      h16, bnscale, bnshift, (size_t)BATCH * FEAT);

  // 4) GEMM2: qk = h @ W_qk^T
  wmma_gemm_kernel<0><<<dim3(QKDIM / 64, BATCH / 128), blk, 0, stream>>>(
      h16, wqk16, nullptr, qk16, nullptr, nullptr, BATCH, QKDIM, FEAT);

  // 5) logits = features @ W_fc^T + b_fc  (N padded to one 16-wide WMMA tile)
  wmma_logits_kernel<<<dim3(BATCH / 128), blk, 0, stream>>>(
      feat16, wfc16, b_fc, logits32, BATCH, FEAT);

  // 6) per-row correlation head -> final output
  attn_head_kernel<<<dim3(BATCH), blk, 0, stream>>>(qk16, logits32, gamma, out);
}
